// GraphDMSABlock_71270687309847
// MI455X (gfx1250) — compile-verified, matrix operand order checked by reading, not run on hardware
//
#include <hip/hip_runtime.h>
#include <hip/hip_bf16.h>
#include <math.h>

// Problem dims (match reference)
#define L_  2
#define H_  4
#define DM_ 128
#define DK_ 32
#define DV_ 32
#define DI_ 256
#define D_  17
#define B_  32
#define T_  128

typedef __attribute__((ext_vector_type(16))) __bf16 v16bf;
typedef __attribute__((ext_vector_type(8)))  float  v8f;

union Frag { unsigned int w[8]; v16bf v; };

// fp32 -> bf16 (round-to-nearest-even), pack two into one dword
__device__ inline unsigned int pk_bf16(float a, float b) {
  unsigned int ua = __float_as_uint(a), ub = __float_as_uint(b);
  ua = (ua + 0x7FFFu + ((ua >> 16) & 1u)) >> 16;
  ub = (ub + 0x7FFFu + ((ub >> 16) & 1u)) >> 16;
  return (ua & 0xFFFFu) | (ub << 16);
}

// CDNA5 16-bit A/B fragment k-pair index for VGPR vv, lane-half hi
// (CDNA5 ISA 7.12.2: lanes0-15 hold K=0..7,16..23 pairs; lanes16-31 K=8..15,24..31)
__device__ inline int kp_map(int vv, int hi) {
  return vv + ((vv >= 4) ? 4 : 0) + (hi ? 4 : 0);
}

__device__ inline v8f wmma_bf16(const Frag& a, const Frag& b, v8f c) {
  return __builtin_amdgcn_wmma_f32_16x16x32_bf16(false, a.v, false, b.v,
                                                 (short)0, c, false, false);
}

// ---------------------------------------------------------------- featproj
__global__ __launch_bounds__(128) void k_featproj(
    const float* __restrict__ xf, const float* __restrict__ fpW,
    const float* __restrict__ fpb, const float* __restrict__ emb,
    float* __restrict__ X) {
  int i = blockIdx.x, c = threadIdx.x;
  int d = i % D_;
  float a = xf[2 * i], b = xf[2 * i + 1];
  X[(size_t)i * DM_ + c] = a * fpW[c] + b * fpW[DM_ + c] + fpb[c] + emb[d * DM_ + c];
}

// ---------------------------------------------------------------- generic GEMM
#define GF_BIAS 1
#define GF_GELU 2
#define GF_RES  4

__global__ __launch_bounds__(256) void k_gemm(
    const float* __restrict__ A, int lda,
    const float* __restrict__ W, int ldw,
    const float* __restrict__ bias,
    const float* __restrict__ resid,
    float* __restrict__ C, int ldc,
    int M, int N, int K, int flags) {
  __shared__ unsigned int aT[64 * 16];   // A tile 64x32 bf16, pair-packed along k
  __shared__ unsigned int bT[16 * 64];   // B tile 32x64 bf16, pair-packed along k
  int tid = threadIdx.x, lane = tid & 31, wv = tid >> 5;
  int ll = lane & 15, hi = lane >> 4;
  int m0 = blockIdx.x * 64, n0 = blockIdx.y * 64;
  int wm = (wv & 3) * 16, wn = (wv >> 2) * 32;
  v8f acc0 = {}, acc1 = {};
  for (int k0 = 0; k0 < K; k0 += 32) {
    __syncthreads();
    for (int i = tid; i < 64 * 16; i += 256) {
      int r = i >> 4, c = (i & 15) * 2;
      const float* p = A + (size_t)(m0 + r) * lda + k0 + c;
      aT[i] = pk_bf16(p[0], p[1]);
    }
    for (int i = tid; i < 16 * 64; i += 256) {
      int kr = (i >> 6) * 2, c = i & 63;
      const float* p = W + (size_t)(k0 + kr) * ldw + n0 + c;
      bT[i] = pk_bf16(p[0], p[ldw]);
    }
    if (k0 + 32 < K)  // global_prefetch next A k-slab
      __builtin_prefetch(A + (size_t)(m0 + (tid >> 2)) * lda + k0 + 32, 0, 1);
    __syncthreads();
    Frag fa, fb0, fb1;
#pragma unroll
    for (int vv = 0; vv < 8; ++vv) {
      int kpi = kp_map(vv, hi);
      fa.w[vv]  = aT[(wm + ll) * 16 + kpi];
      fb0.w[vv] = bT[kpi * 64 + wn + ll];
      fb1.w[vv] = bT[kpi * 64 + wn + 16 + ll];
    }
    acc0 = wmma_bf16(fa, fb0, acc0);
    acc1 = wmma_bf16(fa, fb1, acc1);
  }
#pragma unroll
  for (int r = 0; r < 8; ++r) {
    int row = m0 + wm + r + hi * 8;
#pragma unroll
    for (int half = 0; half < 2; ++half) {
      int col = n0 + wn + half * 16 + ll;
      float x = (half == 0) ? acc0[r] : acc1[r];
      if (flags & GF_BIAS) x += bias[col];
      if (flags & GF_GELU) x = 0.5f * x * (1.0f + erff(x * 0.70710678118654752f));
      if (flags & GF_RES)  x += resid[(size_t)row * ldc + col];
      C[(size_t)row * ldc + col] = x;
    }
  }
}

// ---------------------------------------------------------------- temporal attention
// one block per (b*D + d, h): T=128, head dim 32, diagonal mask, writes attn + ctx
__global__ __launch_bounds__(256) void k_temporal_attn(
    const float* __restrict__ q, const float* __restrict__ k, const float* __restrict__ v,
    float* __restrict__ attn, float* __restrict__ ctx) {
  extern __shared__ char smem[];
  unsigned int* qp  = (unsigned int*)smem;             // [128*16] q pair-packed
  unsigned int* kpp = (unsigned int*)(smem + 8192);    // [128*16] k pair-packed
  unsigned int* vp  = (unsigned int*)(smem + 16384);   // [64*32]  v packed along s
  float*        S   = (float*)(smem + 24576);          // [128*128] scores
  unsigned int* Su  = (unsigned int*)S;                // bf16 a, repacked in place

  int bd = blockIdx.x, h = blockIdx.y;
  int b = bd / D_, d = bd % D_;
  int tid = threadIdx.x, lane = tid & 31, wv = tid >> 5;
  int ll = lane & 15, hi = lane >> 4;
  size_t rstride = (size_t)D_ * DM_;
  size_t base = ((size_t)(b * T_) * D_ + d) * DM_ + h * DK_;

  for (int i = tid; i < 128 * 16; i += 256) {
    int r = i >> 4, c = (i & 15) * 2;
    const float* pq = q + base + (size_t)r * rstride + c;
    const float* pk = k + base + (size_t)r * rstride + c;
    qp[i]  = pk_bf16(pq[0], pq[1]);
    kpp[i] = pk_bf16(pk[0], pk[1]);
  }
  for (int i = tid; i < 64 * 32; i += 256) {
    int kr = (i >> 5) * 2, c = i & 31;
    const float* pv = v + base + (size_t)kr * rstride + c;
    vp[i] = pk_bf16(pv[0], pv[rstride]);
  }
  __syncthreads();

  { // S = q k^T / sqrt(32), diag -> -inf
    int m0 = wv * 16;
    Frag fa;
#pragma unroll
    for (int vv = 0; vv < 8; ++vv) fa.w[vv] = qp[(m0 + ll) * 16 + kp_map(vv, hi)];
    for (int nt = 0; nt < 8; ++nt) {
      int n0 = nt * 16;
      Frag fb;
#pragma unroll
      for (int vv = 0; vv < 8; ++vv) fb.w[vv] = kpp[(n0 + ll) * 16 + kp_map(vv, hi)];
      v8f acc = {};
      acc = wmma_bf16(fa, fb, acc);
#pragma unroll
      for (int r = 0; r < 8; ++r) {
        int row = m0 + r + hi * 8, col = n0 + ll;
        float val = acc[r] * 0.17677669529663687f;  // 1/sqrt(32)
        S[row * 128 + col] = (row == col) ? -1e30f : val;
      }
    }
  }
  __syncthreads();

  if (tid < 128) { // softmax per row; emit attn to global; repack bf16 in place
    int t = tid;
    float mx = -3.0e38f;
    for (int s = 0; s < 128; ++s) mx = fmaxf(mx, S[t * 128 + s]);
    float sum = 0.f;
    for (int s = 0; s < 128; ++s) sum += expf(S[t * 128 + s] - mx);
    float inv = 1.0f / sum;
    float* arow = attn + (((size_t)(bd * H_ + h)) * T_ + t) * T_;
    for (int sp = 0; sp < 64; ++sp) {
      float a0 = expf(S[t * 128 + 2 * sp]     - mx) * inv;
      float a1 = expf(S[t * 128 + 2 * sp + 1] - mx) * inv;
      arow[2 * sp] = a0; arow[2 * sp + 1] = a1;
      Su[t * 128 + sp] = pk_bf16(a0, a1);   // write idx sp <= read idx 2sp: safe
    }
  }
  __syncthreads();

  { // ctx = a @ v  (K=128 -> 4 wmma steps)
    int m0 = wv * 16;
    v8f acc0 = {}, acc1 = {};
    for (int ks = 0; ks < 4; ++ks) {
      Frag fa, fb0, fb1;
#pragma unroll
      for (int vv = 0; vv < 8; ++vv) {
        int kpi = ks * 16 + kp_map(vv, hi);
        fa.w[vv]  = Su[(m0 + ll) * 128 + kpi];
        fb0.w[vv] = vp[kpi * 32 + ll];
        fb1.w[vv] = vp[kpi * 32 + 16 + ll];
      }
      acc0 = wmma_bf16(fa, fb0, acc0);
      acc1 = wmma_bf16(fa, fb1, acc1);
    }
#pragma unroll
    for (int r = 0; r < 8; ++r) {
      int row = m0 + r + hi * 8;
      float* crow = ctx + base + (size_t)row * rstride;
      crow[ll] = acc0[r];
      crow[16 + ll] = acc1[r];
    }
  }
}

// ---------------------------------------------------------------- spatial attention
// one block per n = b*T+t; all 4 heads; D=17 padded to 32; bias P; softmax internal
__global__ __launch_bounds__(256) void k_spatial_attn(
    const float* __restrict__ q, const float* __restrict__ k, const float* __restrict__ v,
    const float* __restrict__ Pb, float* __restrict__ ctx) {
  extern __shared__ char smem[];
  unsigned int* qp  = (unsigned int*)smem;             // [32*64]
  unsigned int* kpp = (unsigned int*)(smem + 8192);    // [32*64]
  unsigned int* vp  = (unsigned int*)(smem + 16384);   // [16*128]
  float*        S   = (float*)(smem + 24576);          // [4*32*32]
  unsigned int* ap  = (unsigned int*)(smem + 40960);   // [4*32*16]

  int n = blockIdx.x;
  int tid = threadIdx.x, lane = tid & 31, wv = tid >> 5;
  int ll = lane & 15, hi = lane >> 4;
  size_t base = (size_t)n * D_ * DM_;

  for (int i = tid; i < 32 * 64; i += 256) {
    int r = i >> 6, c = (i & 63) * 2;
    if (r < D_) {
      const float* pq = q + base + (size_t)r * DM_ + c;
      const float* pk = k + base + (size_t)r * DM_ + c;
      qp[i]  = pk_bf16(pq[0], pq[1]);
      kpp[i] = pk_bf16(pk[0], pk[1]);
    } else { qp[i] = 0u; kpp[i] = 0u; }
  }
  for (int i = tid; i < 16 * 128; i += 256) {
    int kr = (i >> 7) * 2, c = i & 127;
    float v0 = (kr     < D_) ? v[base + (size_t)kr * DM_ + c]       : 0.f;
    float v1 = (kr + 1 < D_) ? v[base + (size_t)(kr + 1) * DM_ + c] : 0.f;
    vp[i] = pk_bf16(v0, v1);
  }
  __syncthreads();

  { // scores per head; wave -> (h, m-tile)
    int h = wv >> 1, m0 = (wv & 1) * 16;
    Frag fa;
#pragma unroll
    for (int vv = 0; vv < 8; ++vv)
      fa.w[vv] = qp[(m0 + ll) * 64 + h * 16 + kp_map(vv, hi)];
    float* Sh = S + h * 1024;
    for (int nt = 0; nt < 2; ++nt) {
      Frag fb;
#pragma unroll
      for (int vv = 0; vv < 8; ++vv)
        fb.w[vv] = kpp[(nt * 16 + ll) * 64 + h * 16 + kp_map(vv, hi)];
      v8f acc = {};
      acc = wmma_bf16(fa, fb, acc);
#pragma unroll
      for (int r = 0; r < 8; ++r) {
        int row = m0 + r + hi * 8, col = nt * 16 + ll;
        float val = acc[r] * 0.17677669529663687f;
        if (row < D_ && col < D_) val += Pb[row * D_ + col];
        Sh[row * 32 + col] = val;
      }
    }
  }
  __syncthreads();

  if (tid < 128) { // softmax over valid 17 cols; pack bf16 a (zero-padded)
    int h = tid >> 5, r = tid & 31;
    float* Sh = S + h * 1024;
    unsigned int* aprow = ap + (h * 32 + r) * 16;
    if (r < D_) {
      float mx = -3.0e38f;
      for (int j = 0; j < D_; ++j) mx = fmaxf(mx, Sh[r * 32 + j]);
      float sum = 0.f;
      for (int j = 0; j < D_; ++j) sum += expf(Sh[r * 32 + j] - mx);
      float inv = 1.0f / sum;
      for (int jp = 0; jp < 16; ++jp) {
        int j0 = 2 * jp, j1 = 2 * jp + 1;
        float a0 = (j0 < D_) ? expf(Sh[r * 32 + j0] - mx) * inv : 0.f;
        float a1 = (j1 < D_) ? expf(Sh[r * 32 + j1] - mx) * inv : 0.f;
        aprow[jp] = pk_bf16(a0, a1);
      }
    } else {
      for (int jp = 0; jp < 16; ++jp) aprow[jp] = 0u;
    }
  }
  __syncthreads();

  { // ctx = a @ v per head (K padded to 32, single wmma step)
    int h = wv >> 1, m0 = (wv & 1) * 16;
    Frag fa, fb0, fb1;
#pragma unroll
    for (int vv = 0; vv < 8; ++vv) {
      int kpi = kp_map(vv, hi);
      fa.w[vv]  = ap[(h * 32 + m0 + ll) * 16 + kpi];
      fb0.w[vv] = vp[kpi * 128 + h * 32 + ll];
      fb1.w[vv] = vp[kpi * 128 + h * 32 + 16 + ll];
    }
    v8f acc0 = {}, acc1 = {};
    acc0 = wmma_bf16(fa, fb0, acc0);
    acc1 = wmma_bf16(fa, fb1, acc1);
#pragma unroll
    for (int r = 0; r < 8; ++r) {
      int row = m0 + r + hi * 8;
      if (row < D_) {
        float* crow = ctx + base + (size_t)row * DM_ + h * 32;
        crow[ll] = acc0[r];
        crow[16 + ll] = acc1[r];
      }
    }
  }
}

// ---------------------------------------------------------------- layernorm (rows of 128)
__global__ __launch_bounds__(128) void k_layernorm(float* __restrict__ X,
    const float* __restrict__ g, const float* __restrict__ b) {
  __shared__ float r1[128], r2[128];
  int row = blockIdx.x, c = threadIdx.x;
  float x = X[(size_t)row * DM_ + c];
  r1[c] = x; r2[c] = x * x;
  __syncthreads();
  for (int s = 64; s > 0; s >>= 1) {
    if (c < s) { r1[c] += r1[c + s]; r2[c] += r2[c + s]; }
    __syncthreads();
  }
  float mean = r1[0] * (1.0f / DM_);
  float var  = r2[0] * (1.0f / DM_) - mean * mean;
  X[(size_t)row * DM_ + c] = (x - mean) * rsqrtf(var + 1e-5f) * g[c] + b[c];
}

// ---------------------------------------------------------------- gated combine
__global__ void k_combine(const float* __restrict__ tb, const float* __restrict__ sb,
                          const float* __restrict__ gate, int l,
                          float* __restrict__ o, int n) {
  float alpha = 1.0f / (1.0f + expf(-gate[l]));
  for (int i = blockIdx.x * blockDim.x + threadIdx.x; i < n; i += gridDim.x * blockDim.x)
    o[i] = (1.0f - alpha) * tb[i] + alpha * sb[i];
}

// ---------------------------------------------------------------- pooled mean + output head
__global__ __launch_bounds__(128) void k_final(const float* __restrict__ X,
    const float* __restrict__ oW, const float* __restrict__ ob,
    float* __restrict__ pooled, float* __restrict__ imp) {
  __shared__ float red[128];
  int n = blockIdx.x, c = threadIdx.x;
  float xv[D_];
  float s = 0.f;
  for (int d = 0; d < D_; ++d) {
    xv[d] = X[((size_t)n * D_ + d) * DM_ + c];
    s += xv[d];
  }
  pooled[(size_t)n * DM_ + c] = s * (1.0f / D_);
  float w = oW[c];
  for (int d = 0; d < D_; ++d) {
    red[c] = xv[d] * w;
    __syncthreads();
    for (int st = 64; st > 0; st >>= 1) { if (c < st) red[c] += red[c + st]; __syncthreads(); }
    if (c == 0) imp[(size_t)n * D_ + d] = red[0] + ob[0];
    __syncthreads();
  }
}

// ---------------------------------------------------------------- host orchestration
extern "C" void kernel_launch(void* const* d_in, const int* in_sizes, int n_in,
                              void* d_out, int out_size, void* d_ws, size_t ws_size,
                              hipStream_t stream) {
  (void)in_sizes; (void)n_in; (void)out_size; (void)ws_size;
  const float* x_feat = (const float*)d_in[0];
  const float* P      = (const float*)d_in[1];
  const float* emb    = (const float*)d_in[2];
  const float* fpW    = (const float*)d_in[3];
  const float* fpb    = (const float*)d_in[4];
  const float* t_Wq   = (const float*)d_in[5];
  const float* t_Wk   = (const float*)d_in[6];
  const float* t_Wv   = (const float*)d_in[7];
  const float* t_Wo   = (const float*)d_in[8];
  const float* t_g    = (const float*)d_in[9];
  const float* t_bb   = (const float*)d_in[10];
  const float* s_Wq   = (const float*)d_in[11];
  const float* s_Wk   = (const float*)d_in[12];
  const float* s_Wv   = (const float*)d_in[13];
  const float* s_Wo   = (const float*)d_in[14];
  const float* s_bo   = (const float*)d_in[15];
  const float* s_g    = (const float*)d_in[16];
  const float* s_bb   = (const float*)d_in[17];
  const float* f_W1   = (const float*)d_in[18];
  const float* f_b1   = (const float*)d_in[19];
  const float* f_W2   = (const float*)d_in[20];
  const float* f_b2   = (const float*)d_in[21];
  const float* f_g    = (const float*)d_in[22];
  const float* f_bb   = (const float*)d_in[23];
  const float* gate   = (const float*)d_in[24];
  const float* out_W  = (const float*)d_in[25];
  const float* out_b  = (const float*)d_in[26];

  float* ws = (float*)d_ws;
  const size_t NX = (size_t)B_ * T_ * D_ * DM_;   // 8,912,896 floats
  float* X    = ws;
  float* Qb   = ws + 1 * NX;
  float* Kb   = ws + 2 * NX;
  float* Vb   = ws + 3 * NX;
  float* CTX  = ws + 4 * NX;
  float* TOUT = ws + 5 * NX;
  float* SOUT = ws + 6 * NX;
  float* HBUF = Kb;     // FFN hidden (2*NX floats) over dead Kb+Vb
  float* COMB = Qb;     // combined input over dead Qb

  float* out      = (float*)d_out;
  float* out_pool = out;
  float* out_imp  = out + (size_t)B_ * T_ * DM_;
  float* out_attn = out_imp + (size_t)B_ * T_ * D_;
  const size_t ATTN_L = (size_t)B_ * D_ * H_ * T_ * T_;

  const int M = B_ * T_ * D_;   // 69632 rows

  auto gemm = [&](const float* A, int lda, const float* W, int ldw,
                  const float* bias, const float* resid,
                  float* C, int ldc, int N, int K, int flags) {
    dim3 g((unsigned)(M / 64), (unsigned)(N / 64));
    k_gemm<<<g, 256, 0, stream>>>(A, lda, W, ldw, bias, resid, C, ldc, M, N, K, flags);
  };

  k_featproj<<<M, 128, 0, stream>>>(x_feat, fpW, fpb, emb, X);

  for (int l = 0; l < L_; ++l) {
    const float* tWq = t_Wq + (size_t)l * DM_ * DM_;
    const float* tWk = t_Wk + (size_t)l * DM_ * DM_;
    const float* tWv = t_Wv + (size_t)l * DM_ * DM_;
    const float* tWo = t_Wo + (size_t)l * DM_ * DM_;
    const float* sWq = s_Wq + (size_t)l * DM_ * DM_;
    const float* sWk = s_Wk + (size_t)l * DM_ * DM_;
    const float* sWv = s_Wv + (size_t)l * DM_ * DM_;
    const float* sWo = s_Wo + (size_t)l * DM_ * DM_;
    const float* fW1 = f_W1 + (size_t)l * DM_ * DI_;
    const float* fW2 = f_W2 + (size_t)l * DI_ * DM_;

    // ---- temporal branch ----
    gemm(X, DM_, tWq, DM_, nullptr, nullptr, Qb, DM_, DM_, DM_, 0);
    gemm(X, DM_, tWk, DM_, nullptr, nullptr, Kb, DM_, DM_, DM_, 0);
    gemm(X, DM_, tWv, DM_, nullptr, nullptr, Vb, DM_, DM_, DM_, 0);
    {
      dim3 g(B_ * D_, H_);
      k_temporal_attn<<<g, 256, 90112, stream>>>(Qb, Kb, Vb,
          out_attn + (size_t)l * ATTN_L, CTX);
    }
    gemm(CTX, DM_, tWo, DM_, nullptr, X, TOUT, DM_, DM_, DM_, GF_RES);
    k_layernorm<<<M, 128, 0, stream>>>(TOUT, t_g + l * DM_, t_bb + l * DM_);

    // ---- spatial branch ----
    gemm(X, DM_, sWq, DM_, nullptr, nullptr, Qb, DM_, DM_, DM_, 0);
    gemm(X, DM_, sWk, DM_, nullptr, nullptr, Kb, DM_, DM_, DM_, 0);
    gemm(X, DM_, sWv, DM_, nullptr, nullptr, Vb, DM_, DM_, DM_, 0);
    k_spatial_attn<<<B_ * T_, 256, 49152, stream>>>(Qb, Kb, Vb, P, CTX);
    gemm(CTX, DM_, sWo, DM_, s_bo + l * DM_, X, SOUT, DM_, DM_, DM_, GF_BIAS | GF_RES);
    k_layernorm<<<M, 128, 0, stream>>>(SOUT, s_g + l * DM_, s_bb + l * DM_);

    // ---- gate + FFN ----
    k_combine<<<2048, 256, 0, stream>>>(TOUT, SOUT, gate, l, COMB, (int)NX);
    gemm(COMB, DM_, fW1, DI_, f_b1 + l * DI_, nullptr, HBUF, DI_, DI_, DM_, GF_BIAS | GF_GELU);
    gemm(HBUF, DI_, fW2, DM_, f_b2 + l * DM_, COMB, X, DM_, DM_, DI_, GF_BIAS | GF_RES);
    k_layernorm<<<M, 128, 0, stream>>>(X, f_g + l * DM_, f_bb + l * DM_);
  }

  k_final<<<B_ * T_, 128, 0, stream>>>(X, out_W, out_b, out_pool, out_imp);
}
